// GCNEncoder_66915590472502
// MI455X (gfx1250) — matrix operand order repaired and checked by reading, not compiled, not run on hardware
//
#include <hip/hip_runtime.h>
#include <hip/hip_bf16.h>
#include <cstdint>
#include <cstddef>

typedef __attribute__((ext_vector_type(16))) _Float16 v16h;
typedef __attribute__((ext_vector_type(8)))  _Float16 v8h;
typedef __attribute__((ext_vector_type(8)))  float    v8f;

#define ATOMIC_ADD_F32(p, v) \
  __hip_atomic_fetch_add((p), (v), __ATOMIC_RELAXED, __HIP_MEMORY_SCOPE_AGENT)

__device__ __forceinline__ v16h cat8(v8h lo, v8h hi) {
  return __builtin_shufflevector(lo, hi, 0, 1, 2, 3, 4, 5, 6, 7,
                                 8, 9, 10, 11, 12, 13, 14, 15);
}

// ---------------------------------------------------------------------------
// degree / normalization
// ---------------------------------------------------------------------------
__global__ void k_fill(float* __restrict__ p, float v, int n) {
  int i = blockIdx.x * blockDim.x + threadIdx.x;
  if (i < n) p[i] = v;
}

__global__ void k_deg_accum(const int* __restrict__ dst, float* __restrict__ deg, int E) {
  int e = blockIdx.x * blockDim.x + threadIdx.x;
  if (e < E) ATOMIC_ADD_F32(&deg[dst[e]], 1.0f);
}

__global__ void k_rsqrt_ip(float* __restrict__ p, int n) {
  int i = blockIdx.x * blockDim.x + threadIdx.x;
  if (i < n) p[i] = rsqrtf(p[i]);  // deg >= 1 always (self-loop)
}

// ---------------------------------------------------------------------------
// WMMA GEMM:  Y[nrows, FOUT] = X[nrows, FIN] @ W[FIN, FOUT]  (f32 in/out,
// f16 operands, f32 accumulate via v_wmma_f32_16x16x32_f16).
//
// Block: 128 threads = 4 waves; block tile = 64 rows; wave tile = 16 rows.
// Whole K (FIN<=128) staged once in LDS.
//
// Fragment loads are arranged to be 16-byte contiguous so they lower to
// ds_load_b128 (2 loads per 8-VGPR fragment):
//   A (16x32, row-major sX): lane needs K = base+{0..7} and base+{16..23}
//     -> two contiguous 8-half chunks at +0 and +16 halves.
//   B (32x16): lane needs K = base+{0..7} and base+{16..23} at a fixed column -> stage W
//     TRANSPOSED (sWt[col][k]) so each 8-half chunk is contiguous.
// Padded strides (136 halves = 272 B) keep every chunk 16B-aligned.
// ---------------------------------------------------------------------------
template <int FIN, int FOUT>
__global__ __launch_bounds__(128) void k_gemm_wmma(const float* __restrict__ X,
                                                   const float* __restrict__ W,
                                                   float* __restrict__ Y,
                                                   int nrows) {
  static_assert(FIN % 32 == 0 && FOUT % 16 == 0, "tile shapes");
  constexpr int ROWS = 64;
  constexpr int LDX  = FIN + 8;  // halves; 272B row stride (16B aligned)
  constexpr int LDK  = FIN + 8;  // halves; stride of transposed W rows
  __shared__ _Float16 sX[ROWS * LDX];
  __shared__ _Float16 sWt[FOUT * LDK];

  const int tid  = threadIdx.x;
  const int wave = tid >> 5;
  const int lane = tid & 31;
  const int row0 = blockIdx.x * ROWS;

  // stage X tile (64 x FIN) as f16, zero-fill out-of-range rows
  for (int i = tid; i < ROWS * FIN; i += 128) {
    int r = i / FIN, c = i % FIN;
    int gr = row0 + r;
    float v = (gr < nrows) ? X[(size_t)gr * FIN + c] : 0.0f;
    sX[r * LDX + c] = (_Float16)v;
  }
  // stage W (FIN x FOUT) transposed: sWt[col * LDK + k]
  for (int i = tid; i < FIN * FOUT; i += 128) {
    int k = i / FOUT, c = i % FOUT;
    sWt[c * LDK + k] = (_Float16)W[i];
  }
  __syncthreads();

  const int halfsel = lane >> 4;          // 0: lanes 0-15, 1: lanes 16-31
  const int col16   = lane & 15;
  const int arow    = wave * 16 + col16;  // A row for this lane
  const int akadd   = halfsel << 3;       // A: upper half-lanes at K+8
  const int bkadd   = halfsel << 3;       // B: upper half-lanes at K+8

  constexpr int NT = FOUT / 16;
  v8f acc[NT];
#pragma unroll
  for (int t = 0; t < NT; ++t) acc[t] = v8f{};

#pragma unroll
  for (int k0 = 0; k0 < FIN; k0 += 32) {
    // A fragment: two contiguous 16B chunks (K base+0..7, base+16..23)
    const _Float16* ax = sX + arow * LDX + k0 + akadd;
    v16h a = cat8(*(const v8h*)ax, *(const v8h*)(ax + 16));
#pragma unroll
    for (int t = 0; t < NT; ++t) {
      // B fragment: two 16B chunks at column (t*16 + col16) (K base+0..7, base+16..23)
      const _Float16* bx = sWt + (t * 16 + col16) * LDK + k0 + bkadd;
      v16h b = cat8(*(const v8h*)bx, *(const v8h*)(bx + 16));
      acc[t] = __builtin_amdgcn_wmma_f32_16x16x32_f16(
          false, a, false, b, (short)0, acc[t], false, false);
    }
  }

  // D tile store: VGPR v, lane L -> row = (L>=16 ? 8 : 0)+v, col = L&15
  const int rbase = row0 + wave * 16 + (halfsel << 3);
#pragma unroll
  for (int t = 0; t < NT; ++t) {
    const int col = t * 16 + col16;
#pragma unroll
    for (int v = 0; v < 8; ++v) {
      int gr = rbase + v;
      if (gr < nrows) Y[(size_t)gr * FOUT + col] = acc[t][v];
    }
  }
}

// ---------------------------------------------------------------------------
// out[i,f] = H[i,f] * dinv[i]^2 + bias[f]   (self-loop term + bias; fully
// initializes the destination so subsequent atomics start clean)
// ---------------------------------------------------------------------------
__global__ void k_selfloop_bias(const float* __restrict__ H,
                                const float* __restrict__ dinv,
                                const float* __restrict__ bias,
                                float* __restrict__ out, int N, int F) {
  int i = blockIdx.x * blockDim.x + threadIdx.x;
  int n = i / F, f = i % F;
  if (n < N) {
    float di = dinv[n];
    out[i] = H[i] * di * di + bias[f];
  }
}

// ---------------------------------------------------------------------------
// per-edge gather + scaled atomic scatter-add:
//   agg[dst, :] += H[src, :] * dinv[src] * dinv[dst]
// One thread handles one (edge, 4-feature) chunk; a wave covers one edge's
// feature row coalesced (float4 gather, 4x global_atomic_add_f32).
// ---------------------------------------------------------------------------
template <int F>
__global__ void k_edge_scatter(const float* __restrict__ H,
                               const int* __restrict__ src,
                               const int* __restrict__ dst,
                               const float* __restrict__ dinv,
                               float* __restrict__ agg, int E) {
  constexpr int Q = F / 4;
  long long tid = (long long)blockIdx.x * blockDim.x + threadIdx.x;
  if (tid >= (long long)E * Q) return;
  int e  = (int)(tid / Q);
  int fq = (int)(tid % Q) * 4;
  int s = src[e], d = dst[e];
  float nrm = dinv[s] * dinv[d];
  const float4 v = *(const float4*)(H + (size_t)s * F + fq);
  float* base = agg + (size_t)d * F + fq;
  ATOMIC_ADD_F32(base + 0, v.x * nrm);
  ATOMIC_ADD_F32(base + 1, v.y * nrm);
  ATOMIC_ADD_F32(base + 2, v.z * nrm);
  ATOMIC_ADD_F32(base + 3, v.w * nrm);
}

__global__ void k_relu_ip(float* __restrict__ p, int n) {
  int i = blockIdx.x * blockDim.x + threadIdx.x;
  if (i < n) p[i] = fmaxf(p[i], 0.0f);
}

// ---------------------------------------------------------------------------
// launch
// ---------------------------------------------------------------------------
static inline unsigned nblk(long long n, int t) {
  return (unsigned)((n + t - 1) / t);
}

extern "C" void kernel_launch(void* const* d_in, const int* in_sizes, int n_in,
                              void* d_out, int out_size, void* d_ws, size_t ws_size,
                              hipStream_t stream) {
  const float* x  = (const float*)d_in[0];
  const int*   ei = (const int*)d_in[1];
  const float* W1 = (const float*)d_in[2];
  const float* b1 = (const float*)d_in[3];
  const float* W2 = (const float*)d_in[4];
  const float* b2 = (const float*)d_in[5];
  float*       out = (float*)d_out;

  constexpr int FIN = 128, HID = 128, FOUT = 64;
  const int N = in_sizes[0] / FIN;
  const int E = in_sizes[1] / 2;
  const int* src = ei;
  const int* dst = ei + E;

  // workspace layout (floats): dinv[N] | H[N*HID] (reused for H2) | AGG1[N*HID]
  float* dinv = (float*)d_ws;
  float* H    = dinv + N;
  float* AGG1 = H + (size_t)N * HID;

  // --- normalization: deg -> dinv ---
  k_fill<<<nblk(N, 256), 256, 0, stream>>>(dinv, 1.0f, N);
  k_deg_accum<<<nblk(E, 256), 256, 0, stream>>>(dst, dinv, E);
  k_rsqrt_ip<<<nblk(N, 256), 256, 0, stream>>>(dinv, N);

  // --- layer 1: H = x @ W1 ; AGG1 = selfloop + b1 + edge scatter ; relu ---
  k_gemm_wmma<FIN, HID><<<nblk(N, 64), 128, 0, stream>>>(x, W1, H, N);
  k_selfloop_bias<<<nblk((long long)N * HID, 256), 256, 0, stream>>>(
      H, dinv, b1, AGG1, N, HID);
  k_edge_scatter<HID><<<nblk((long long)E * (HID / 4), 256), 256, 0, stream>>>(
      H, src, dst, dinv, AGG1, E);
  k_relu_ip<<<nblk((long long)N * HID, 256), 256, 0, stream>>>(AGG1, N * HID);

  // --- layer 2: H2 = AGG1 @ W2 (into H buffer) ; out = selfloop + b2 + scatter ---
  k_gemm_wmma<HID, FOUT><<<nblk(N, 64), 128, 0, stream>>>(AGG1, W2, H, N);
  k_selfloop_bias<<<nblk((long long)N * FOUT, 256), 256, 0, stream>>>(
      H, dinv, b2, out, N, FOUT);
  k_edge_scatter<FOUT><<<nblk((long long)E * (FOUT / 4), 256), 256, 0, stream>>>(
      H, src, dst, dinv, out, E);
}
